// IntegrableMLP_30382598652038
// MI455X (gfx1250) — matrix operand order repaired and checked by reading, not compiled, not run on hardware
//
#include <hip/hip_runtime.h>
#include <hip/hip_bf16.h>

// MI455X / gfx1250 forward-mode Jacobian of a 2->32->32->16 swish MLP.
// Compute-bound (AI ~600 flop/byte): all K=32 matmuls mapped to
// v_wmma_f32_16x16x32_f16. One wave32 per 16-row batch tile, 9 WMMAs/tile.
// Sigmoid via CDNA5 v_tanh_f32: 1 trans op instead of exp+rcp (2).

typedef __attribute__((ext_vector_type(16))) _Float16 v16h;
typedef __attribute__((ext_vector_type(8)))  _Float16 v8h;
typedef __attribute__((ext_vector_type(8)))  float    v8f;

#define WAVES_PER_BLOCK 8
#define ACT_STRIDE 40   // halves per LDS row (80 B): rows 16B-aligned, banks spread

__device__ __forceinline__ float fast_sigmoid(float z) {
    // sigmoid(z) = 0.5*tanh(z/2) + 0.5  ->  single v_tanh_f32 (CDNA5 TRANS op)
    float t;
#if __has_builtin(__builtin_amdgcn_tanhf)
    t = __builtin_amdgcn_tanhf(z * 0.5f);
#else
    const float h = z * 0.5f;
    asm("v_tanh_f32 %0, %1" : "=v"(t) : "v"(h));
#endif
    return fmaf(0.5f, t, 0.5f);
}

__global__ __launch_bounds__(256) void mlp_jac_wmma(
    const float* __restrict__ x,
    const float* __restrict__ W0, const float* __restrict__ b0,
    const float* __restrict__ W1, const float* __restrict__ b1,
    const float* __restrict__ W2, const float* __restrict__ b2,
    const float* __restrict__ V,
    float* __restrict__ out, int ntiles)
{
    __shared__ __align__(16) _Float16 ldsAct[WAVES_PER_BLOCK][16 * ACT_STRIDE];
    __shared__ float ldsOut[WAVES_PER_BLOCK][32];

    const int lane   = threadIdx.x & 31;
    const int wib    = threadIdx.x >> 5;
    const int gwave  = (int)((blockIdx.x * blockDim.x + threadIdx.x) >> 5);
    const int nwaves = (int)((gridDim.x * blockDim.x) >> 5);

    _Float16* act  = ldsAct[wib];
    float*    outS = ldsOut[wib];

    const int half = lane >> 4;      // which 16-lane half
    const int lm   = lane & 15;      // position within half
    const int kb   = half * 8;       // A-fragment K base for this half

    // ---- Layer-0 per-lane constants in A-fragment element order ----
    // A-frag (f16 16x32): lane half<16 element j -> K = (j<8 ? kb+j : 16+kb+j-8)
    float w00[16], w01[16], bb0[16];
#pragma unroll
    for (int j = 0; j < 16; ++j) {
        const int o = (j < 8) ? (kb + j) : (16 + kb + (j - 8));
        w00[j] = W0[o * 2 + 0];
        w01[j] = W0[o * 2 + 1];
        bb0[j] = b0[o];
    }
    // ---- B fragments (32x16, K x N), B[k][n] = W[n][k] ----
    // lane: n = lm (+16 for second N-tile), element j -> K = (j<8 ? kb+j : 16+kb+j-8)
    v16h Bw1a, Bw1b, Bw2;
#pragma unroll
    for (int j = 0; j < 16; ++j) {
        const int k = (j < 8) ? (kb + j) : (16 + kb + (j - 8));
        Bw1a[j] = (_Float16)W1[(lm +  0) * 32 + k];
        Bw1b[j] = (_Float16)W1[(lm + 16) * 32 + k];
        Bw2[j]  = (_Float16)W2[lm * 32 + k];
    }
    const float b1a = b1[lm], b1b = b1[lm + 16], b2v = b2[lm], vv = V[lm];
    const v8f zero8 = {};

    for (int tile = gwave; tile < ntiles; tile += nwaves) {
        const int nextTile = tile + nwaves;
        if (nextTile < ntiles)
            __builtin_prefetch(&x[(nextTile * 16 + lm) * 2], 0, 0);

        const int row = tile * 16 + lm;
        const float x0 = x[row * 2 + 0];
        const float x1 = x[row * 2 + 1];

        // ---- Layer 0 (K=2) computed directly in A-fragment layout ----
        v16h h1f, a0f, a1f;
#pragma unroll
        for (int j = 0; j < 16; ++j) {
            const float z  = fmaf(x0, w00[j], fmaf(x1, w01[j], bb0[j]));
            const float s  = fast_sigmoid(z);
            const float sw = z * s;
            const float d  = sw + s * (1.0f - sw);
            h1f[j] = (_Float16)sw;
            a0f[j] = (_Float16)(d * w00[j]);   // alpha after L0, tangent dim 0
            a1f[j] = (_Float16)(d * w01[j]);   // tangent dim 1
        }

        // ---- Layer 1 forward: z2[16x32] = h1 @ W1^T + b1 (2 WMMAs) ----
        v8f cza, czb;
#pragma unroll
        for (int r = 0; r < 8; ++r) { cza[r] = b1a; czb[r] = b1b; }
        v8f z2a = __builtin_amdgcn_wmma_f32_16x16x32_f16(false, h1f, false, Bw1a, (short)0, cza, false, false);
        v8f z2b = __builtin_amdgcn_wmma_f32_16x16x32_f16(false, h1f, false, Bw1b, (short)0, czb, false, false);

        v8f d2a, d2b;
#pragma unroll
        for (int r = 0; r < 8; ++r) {
            const int m = r + 8 * half;          // C/D row for this register
            {
                const float z = z2a[r], s = fast_sigmoid(z), sw = z * s;
                d2a[r] = sw + s * (1.0f - sw);
                act[m * ACT_STRIDE + lm] = (_Float16)sw;
            }
            {
                const float z = z2b[r], s = fast_sigmoid(z), sw = z * s;
                d2b[r] = sw + s * (1.0f - sw);
                act[m * ACT_STRIDE + 16 + lm] = (_Float16)sw;
            }
        }
        asm volatile("s_wait_dscnt 0" ::: "memory");
        // h2 A-fragment: two 16B LDS loads per lane
        v8h lo = *(const v8h*)&act[lm * ACT_STRIDE + kb];
        v8h hi = *(const v8h*)&act[lm * ACT_STRIDE + 16 + kb];
        v16h h2f = __builtin_shufflevector(lo, hi, 0,1,2,3,4,5,6,7,8,9,10,11,12,13,14,15);

        // ---- Layer 2 forward: z3[16x16] = h2 @ W2^T + b2 (1 WMMA) ----
        v8f cz3;
#pragma unroll
        for (int r = 0; r < 8; ++r) cz3[r] = b2v;
        v8f z3 = __builtin_amdgcn_wmma_f32_16x16x32_f16(false, h2f, false, Bw2, (short)0, cz3, false, false);
        v8f d3;
#pragma unroll
        for (int r = 0; r < 8; ++r) {
            const float z = z3[r], s = fast_sigmoid(z), sw = z * s;
            d3[r] = sw + s * (1.0f - sw);
        }

        // ---- Jacobian streams: beta = alpha @ W^T, alpha = d * beta ----
#pragma unroll
        for (int i = 0; i < 2; ++i) {
            const v16h af = (i == 0) ? a0f : a1f;
            v8f ba = __builtin_amdgcn_wmma_f32_16x16x32_f16(false, af, false, Bw1a, (short)0, zero8, false, false);
            v8f bb = __builtin_amdgcn_wmma_f32_16x16x32_f16(false, af, false, Bw1b, (short)0, zero8, false, false);
#pragma unroll
            for (int r = 0; r < 8; ++r) {
                const int m = r + 8 * half;
                act[m * ACT_STRIDE + lm]      = (_Float16)(d2a[r] * ba[r]);
                act[m * ACT_STRIDE + 16 + lm] = (_Float16)(d2b[r] * bb[r]);
            }
            asm volatile("s_wait_dscnt 0" ::: "memory");
            v8h l2 = *(const v8h*)&act[lm * ACT_STRIDE + kb];
            v8h h2 = *(const v8h*)&act[lm * ACT_STRIDE + 16 + kb];
            v16h a2f = __builtin_shufflevector(l2, h2, 0,1,2,3,4,5,6,7,8,9,10,11,12,13,14,15);
            v8f b3 = __builtin_amdgcn_wmma_f32_16x16x32_f16(false, a2f, false, Bw2, (short)0, zero8, false, false);

            // out[m][i] = sum_n d3*beta3[m][n]*V[n]: butterfly within 16-lane half
#pragma unroll
            for (int r = 0; r < 8; ++r) {
                float g = b3[r] * d3[r] * vv;
                g += __shfl_xor(g, 1, 32);
                g += __shfl_xor(g, 2, 32);
                g += __shfl_xor(g, 4, 32);
                g += __shfl_xor(g, 8, 32);
                if (lm == 0) outS[(r + 8 * half) * 2 + i] = g;   // lanes 0 and 16
            }
        }
        asm volatile("s_wait_dscnt 0" ::: "memory");
        // out tile is 16 rows x 2 = 32 contiguous floats -> one coalesced 128B store
        out[tile * 32 + lane] = outS[lane];
    }
}

extern "C" void kernel_launch(void* const* d_in, const int* in_sizes, int n_in,
                              void* d_out, int out_size, void* d_ws, size_t ws_size,
                              hipStream_t stream) {
    const float* x  = (const float*)d_in[0];
    const float* W0 = (const float*)d_in[1];
    const float* b0 = (const float*)d_in[2];
    const float* W1 = (const float*)d_in[3];
    const float* b1 = (const float*)d_in[4];
    const float* W2 = (const float*)d_in[5];
    const float* b2 = (const float*)d_in[6];
    const float* V  = (const float*)d_in[7];
    float* out = (float*)d_out;

    const int batch  = in_sizes[0] / 2;   // [B, 2] inputs
    const int ntiles = batch / 16;        // 65536 for B = 1M

    dim3 grid(4096), block(256);          // 32768 waves -> 2 tiles per wave
    mlp_jac_wmma<<<grid, block, 0, stream>>>(x, W0, b0, W1, b1, W2, b2, V, out, ntiles);
}
